// ImprovedQuantumEncoder_52639119180158
// MI455X (gfx1250) — compile-verified, matrix operand order checked by reading, not run on hardware
//
// CDNA5 (gfx1250) implementation of ImprovedQuantumEncoder forward.
// All matmul stages lower to v_wmma_f32_16x16x32_bf16 (fp32 accumulate).
// B=8192, IN=512, N=16, D=128, O=256, H=4, HD=32.
#include <hip/hip_runtime.h>
#include <math.h>

typedef __attribute__((ext_vector_type(16))) __bf16 bf16x16;
typedef __attribute__((ext_vector_type(8)))  float  f32x8;

#define WB   8192
#define WIN  512
#define WN   16
#define WD   128
#define WO   256
#define KBIG (1 << 28)

// ---------------------------------------------------------------- WMMA core
__device__ __forceinline__ f32x8 wmma_bf16(bf16x16 a, bf16x16 b, f32x8 c) {
  return __builtin_amdgcn_wmma_f32_16x16x32_bf16(false, a, false, b, (short)0, c,
                                                 false, false);
}

// A fragment (16xK32): lane row = lane&15, K offset = (lane>>4)*8 per ISA layout.
__device__ __forceinline__ bf16x16 load_a(const float* __restrict__ src, int ld,
                                          int m0, int k0, int kmax, float scale) {
  const int l = threadIdx.x & 31;
  const int m = l & 15, kof = (l >> 4) << 3;
  const float* row = src + (size_t)(m0 + m) * ld;
  bf16x16 a;
#pragma unroll
  for (int j = 0; j < 8; ++j) {
    int k1 = k0 + kof + j, k2 = k0 + 16 + kof + j;
    a[j]     = (k1 < kmax) ? (__bf16)(row[k1] * scale) : (__bf16)0.0f;
    a[8 + j] = (k2 < kmax) ? (__bf16)(row[k2] * scale) : (__bf16)0.0f;
  }
  return a;
}

// B fragment (K32x16) from row-major K x N source.
__device__ __forceinline__ bf16x16 load_b(const float* __restrict__ src, int ld,
                                          int k0, int n0, int kmax, float scale) {
  const int l = threadIdx.x & 31;
  const int n = l & 15, kof = (l >> 4) << 3;
  bf16x16 b;
#pragma unroll
  for (int j = 0; j < 8; ++j) {
    int k1 = k0 + kof + j, k2 = k0 + 16 + kof + j;
    b[j]     = (k1 < kmax) ? (__bf16)(src[(size_t)k1 * ld + n0 + n] * scale) : (__bf16)0.0f;
    b[8 + j] = (k2 < kmax) ? (__bf16)(src[(size_t)k2 * ld + n0 + n] * scale) : (__bf16)0.0f;
  }
  return b;
}

// B fragment where B = S^T, S row-major N x K (e.g. K^T in attention scores).
__device__ __forceinline__ bf16x16 load_bT(const float* __restrict__ S, int ld,
                                           int k0, int n0, int kmax, float scale) {
  const int l = threadIdx.x & 31;
  const int n = l & 15, kof = (l >> 4) << 3;
  const float* row = S + (size_t)(n0 + n) * ld;
  bf16x16 b;
#pragma unroll
  for (int j = 0; j < 8; ++j) {
    int k1 = k0 + kof + j, k2 = k0 + 16 + kof + j;
    b[j]     = (k1 < kmax) ? (__bf16)(row[k1] * scale) : (__bf16)0.0f;
    b[8 + j] = (k2 < kmax) ? (__bf16)(row[k2] * scale) : (__bf16)0.0f;
  }
  return b;
}

// bf16-source variants (no convert needed for operand elements).
__device__ __forceinline__ bf16x16 load_a_bf(const __bf16* __restrict__ src, int ld,
                                             int m0, int k0, float scale) {
  const int l = threadIdx.x & 31;
  const int m = l & 15, kof = (l >> 4) << 3;
  const __bf16* row = src + (size_t)(m0 + m) * ld;
  bf16x16 a;
#pragma unroll
  for (int j = 0; j < 8; ++j) {
    a[j]     = (scale == 1.0f) ? row[k0 + kof + j] : (__bf16)((float)row[k0 + kof + j] * scale);
    a[8 + j] = (scale == 1.0f) ? row[k0 + 16 + kof + j]
                               : (__bf16)((float)row[k0 + 16 + kof + j] * scale);
  }
  return a;
}

__device__ __forceinline__ bf16x16 load_bT_bf(const __bf16* __restrict__ S, int ld,
                                              int k0, int n0) {
  const int l = threadIdx.x & 31;
  const int n = l & 15, kof = (l >> 4) << 3;
  const __bf16* row = S + (size_t)(n0 + n) * ld;
  bf16x16 b;
#pragma unroll
  for (int j = 0; j < 8; ++j) {
    b[j]     = row[k0 + kof + j];
    b[8 + j] = row[k0 + 16 + kof + j];
  }
  return b;
}

// B fragment, K limited to 16 rows (upper half zero-padded), bf16 source.
__device__ __forceinline__ bf16x16 load_b_bf_k16(const __bf16* __restrict__ src,
                                                 int ld, int n0) {
  const int l = threadIdx.x & 31;
  const int n = l & 15, kof = (l >> 4) << 3;
  bf16x16 b;
#pragma unroll
  for (int j = 0; j < 8; ++j) {
    b[j]     = src[(size_t)(kof + j) * ld + n0 + n];
    b[8 + j] = (__bf16)0.0f;
  }
  return b;
}

// B fragment K=16 with per-row (per-k) fp32 scale (fold 1/norm into operand).
__device__ __forceinline__ bf16x16 load_b_rows16(const float* __restrict__ src, int ld,
                                                 int n0, const float* __restrict__ rs) {
  const int l = threadIdx.x & 31;
  const int n = l & 15, kof = (l >> 4) << 3;
  bf16x16 b;
#pragma unroll
  for (int j = 0; j < 8; ++j) {
    int k = kof + j;  // 0..15
    b[j]     = (__bf16)(src[(size_t)k * ld + n0 + n] * rs[k]);
    b[8 + j] = (__bf16)0.0f;
  }
  return b;
}

// C/D tile store: n = lane&15, m = r + 8*(lane>>4).
__device__ __forceinline__ void store_c(float* __restrict__ dst, int ld, int m0,
                                        int n0, f32x8 c) {
  const int l = threadIdx.x & 31;
  const int n = l & 15, mb = m0 + ((l >> 4) << 3);
#pragma unroll
  for (int r = 0; r < 8; ++r) dst[(size_t)(mb + r) * ld + n0 + n] = c[r];
}

__device__ __forceinline__ void store_c_bf(__bf16* __restrict__ dst, int ld, int m0,
                                           int n0, f32x8 c) {
  const int l = threadIdx.x & 31;
  const int n = l & 15, mb = m0 + ((l >> 4) << 3);
#pragma unroll
  for (int r = 0; r < 8; ++r) dst[(size_t)(mb + r) * ld + n0 + n] = (__bf16)c[r];
}

__device__ __forceinline__ int gwave() {
  return (int)((blockIdx.x * blockDim.x + threadIdx.x) >> 5);
}

// ---------------------------------------------------------------- kernels
__global__ void k_init(float* magsum, float* pur, float* ent, float* phv) {
  int t = blockIdx.x * blockDim.x + threadIdx.x;
  if (t < WB * WN) magsum[t] = 0.f;
  if (t == 0) { *pur = 0.f; *ent = 0.f; *phv = 0.f; }
}

// Cayley: U = (I - iA)^-1 (I + iA), A = P - P^T, 16x16 complex Gauss-Jordan.
__global__ void k_cayley(const float* __restrict__ Ap, float* Ur, float* Ui) {
  __shared__ float Mr[16][16], Mi[16][16], Rr[16][16], Ri[16][16];
  __shared__ float pr, pi, fr_[16], fi_[16];
  const int t = threadIdx.x;
  for (int idx = t; idx < 256; idx += 32) {
    int i = idx >> 4, j = idx & 15;
    float a = Ap[i * 16 + j] - Ap[j * 16 + i];
    Mr[i][j] = (i == j) ? 1.f : 0.f;  Mi[i][j] = -a;
    Rr[i][j] = (i == j) ? 1.f : 0.f;  Ri[i][j] =  a;
  }
  __syncthreads();
  for (int k = 0; k < 16; ++k) {
    if (t == 0) {
      float dr = Mr[k][k], di = Mi[k][k], inv = 1.f / (dr * dr + di * di);
      pr = dr * inv; pi = -di * inv;
    }
    __syncthreads();
    if (t < 16) {
      float mr = Mr[k][t], mi = Mi[k][t];
      Mr[k][t] = mr * pr - mi * pi;  Mi[k][t] = mr * pi + mi * pr;
      float rr = Rr[k][t], ri = Ri[k][t];
      Rr[k][t] = rr * pr - ri * pi;  Ri[k][t] = rr * pi + ri * pr;
    }
    __syncthreads();
    if (t < 16) { fr_[t] = Mr[t][k]; fi_[t] = Mi[t][k]; }
    __syncthreads();
    for (int idx = t; idx < 256; idx += 32) {
      int i = idx >> 4, j = idx & 15;
      if (i != k) {
        float fr = fr_[i], fi = fi_[i];
        float a = Mr[k][j], b = Mi[k][j];
        Mr[i][j] -= fr * a - fi * b;  Mi[i][j] -= fr * b + fi * a;
        float c = Rr[k][j], d = Ri[k][j];
        Rr[i][j] -= fr * c - fi * d;  Ri[i][j] -= fr * d + fi * c;
      }
    }
    __syncthreads();
  }
  for (int idx = t; idx < 256; idx += 32) {
    Ur[idx] = Rr[idx >> 4][idx & 15];
    Ui[idx] = Ri[idx >> 4][idx & 15];
  }
}

// phases = tanh(x @ Wph + bph) * pi; also accumulate mean per-b variance (ddof=1).
__global__ void k_phases(const float* __restrict__ x, const float* __restrict__ Wph,
                         const float* __restrict__ bph, float* __restrict__ phases,
                         float* __restrict__ d_phvar) {
  const int b0 = gwave() * 16;
  const int l = threadIdx.x & 31, col = l & 15, hi = l >> 4, w = threadIdx.x >> 5;
  f32x8 acc = {};
  for (int ks = 0; ks < WIN; ks += 32) {
    bf16x16 a = load_a(x, WIN, b0, ks, KBIG, 1.f);
    bf16x16 b = load_b(Wph, WN, ks, 0, KBIG, 1.f);
    acc = wmma_bf16(a, b, acc);
  }
  const float bias = bph[col];
  __shared__ float tile[8][16][16];
  float vals[8];
#pragma unroll
  for (int r = 0; r < 8; ++r) {
    vals[r] = tanhf(acc[r] + bias) * 3.14159265358979323846f;
    phases[(size_t)(b0 + 8 * hi + r) * WN + col] = vals[r];
    tile[w][8 * hi + r][col] = vals[r];
  }
  __syncthreads();
  if (l < 16) {
    float mean = 0.f;
#pragma unroll
    for (int j = 0; j < 16; ++j) mean += tile[w][l][j];
    mean *= (1.f / 16.f);
    float v = 0.f;
#pragma unroll
    for (int j = 0; j < 16; ++j) { float d = tile[w][l][j] - mean; v += d * d; }
    atomicAdd(d_phvar, v * (1.f / 15.f) * (1.f / (float)WB));
  }
}

// amp GEMMs + phase rotation -> unnormalized state s_r, s_i (B, N*D).
__global__ void k_encode(const float* __restrict__ x, const float* __restrict__ Wr,
                         const float* __restrict__ br, const float* __restrict__ Wi,
                         const float* __restrict__ bi, const float* __restrict__ phases,
                         float* __restrict__ sr, float* __restrict__ si) {
  const int gw = gwave();
  const int bt = gw >> 7, jt = gw & 127;
  const int b0 = bt * 16, nd0 = jt * 16, n = nd0 >> 7;
  const int l = threadIdx.x & 31, col = l & 15, hi = l >> 4, w = threadIdx.x >> 5;
  f32x8 cr = {}, ci = {};
  for (int ks = 0; ks < WIN; ks += 32) {
    bf16x16 ax = load_a(x, WIN, b0, ks, KBIG, 1.f);
    cr = wmma_bf16(ax, load_b(Wr, WN * WD, ks, nd0, KBIG, 1.f), cr);
    ci = wmma_bf16(ax, load_b(Wi, WN * WD, ks, nd0, KBIG, 1.f), ci);
  }
  __shared__ float cs[8][16][2];
  if (l < 16) {
    float p = phases[(size_t)(b0 + l) * WN + n], sv, cv;
    sincosf(p, &sv, &cv);
    cs[w][l][0] = cv; cs[w][l][1] = sv;
  }
  __syncthreads();
  const float vbr = br[nd0 + col], vbi = bi[nd0 + col];
#pragma unroll
  for (int r = 0; r < 8; ++r) {
    int row = 8 * hi + r;
    float A = cr[r] + vbr, Bv = ci[r] + vbi;
    float cv = cs[w][row][0], sv = cs[w][row][1];
    size_t idx = (size_t)(b0 + row) * (WN * WD) + nd0 + col;
    sr[idx] = A * cv - Bv * sv;
    si[idx] = A * sv + Bv * cv;
  }
}

__global__ void k_norms(const float* __restrict__ sr, const float* __restrict__ si,
                        float* __restrict__ invn) {
  const int bn = gwave();
  const int l = threadIdx.x & 31;
  const float* pr = sr + (size_t)bn * WD;
  const float* pi = si + (size_t)bn * WD;
  float s = 0.f;
  for (int d = l; d < WD; d += 32) { float a = pr[d], b = pi[d]; s += a * a + b * b; }
  for (int off = 16; off > 0; off >>= 1) s += __shfl_xor(s, off);
  if (l == 0) invn[bn] = rsqrtf(s + 1e-8f);
}

// interference: out[n][d] = sum_m (s/norm)[m][d] * U[m][n] (complex), K=16 pad 32.
__global__ void k_interf(const float* __restrict__ Ur, const float* __restrict__ Ui,
                         const float* __restrict__ sr, const float* __restrict__ si,
                         const float* __restrict__ invn,
                         float* __restrict__ outr, float* __restrict__ outi) {
  const int gw = gwave();
  const int b = gw >> 3, d0 = (gw & 7) * 16;
  bf16x16 AUr  = load_b(Ur, WN, 0, 0, 16, 1.f);   // A[n][m] = Ur[m][n]
  bf16x16 AUi  = load_b(Ui, WN, 0, 0, 16, 1.f);
  bf16x16 AnUi = load_b(Ui, WN, 0, 0, 16, -1.f);
  const float* rs = invn + b * WN;
  bf16x16 Bsr = load_b_rows16(sr + (size_t)b * WN * WD, WD, d0, rs);
  bf16x16 Bsi = load_b_rows16(si + (size_t)b * WN * WD, WD, d0, rs);
  f32x8 cr = {}, ci = {};
  cr = wmma_bf16(AUr, Bsr, cr);   cr = wmma_bf16(AnUi, Bsi, cr);
  ci = wmma_bf16(AUi, Bsr, ci);   ci = wmma_bf16(AUr, Bsi, ci);
  store_c(outr + (size_t)b * WN * WD, WD, 0, d0, cr);
  store_c(outi + (size_t)b * WN * WD, WD, 0, d0, ci);
}

__global__ void k_state_stats(const float* __restrict__ sr, const float* __restrict__ si,
                              float* __restrict__ fninv, float* __restrict__ pbn) {
  const int bn = gwave();
  const int l = threadIdx.x & 31;
  const float* pr = sr + (size_t)bn * WD;
  const float* pi = si + (size_t)bn * WD;
  float s1 = 0.f, s2 = 0.f;
  for (int d = l; d < WD; d += 32) {
    float a = pr[d], b = pi[d], m = a * a + b * b;
    s1 += m; s2 += m * m;
  }
  for (int off = 16; off > 0; off >>= 1) { s1 += __shfl_xor(s1, off); s2 += __shfl_xor(s2, off); }
  if (l == 0) {
    fninv[bn] = rsqrtf(s1 + 1e-8f);
    pbn[bn]   = s2 / (s1 * s1 + 1e-8f);
  }
}

__global__ void k_purity_final(const float* __restrict__ pbn, float* pur, float* ent) {
  int b = blockIdx.x * blockDim.x + threadIdx.x;
  float s = 0.f;
#pragma unroll
  for (int n = 0; n < 16; ++n) s += pbn[b * 16 + n];
  float pb = s * (1.f / 16.f);
  atomicAdd(pur, pb * (1.f / (float)WB));
  atomicAdd(ent, -logf(pb + 1e-8f) * (1.f / (float)WB));
}

// fidelity(b,i,j) = clip(|<psi_i|psi_j>|^2, 0, 1): R,I Gram matrices via WMMA.
__global__ void k_fidelity(const float* __restrict__ sr, const float* __restrict__ si,
                           const float* __restrict__ fninv, float* __restrict__ fid) {
  const int b = gwave();
  const int l = threadIdx.x & 31, col = l & 15;
  const float scl = fninv[b * WN + col];           // row scale (A) == col scale (B^T)
  const float* br_ = sr + (size_t)b * WN * WD;
  const float* bi_ = si + (size_t)b * WN * WD;
  f32x8 R = {}, I = {};
  for (int ks = 0; ks < WD; ks += 32) {
    bf16x16 Anr  = load_a (br_, WD, 0, ks, KBIG,  scl);
    bf16x16 Ani  = load_a (bi_, WD, 0, ks, KBIG,  scl);
    bf16x16 Anin = load_a (bi_, WD, 0, ks, KBIG, -scl);
    bf16x16 BnrT = load_bT(br_, WD, ks, 0, KBIG,  scl);
    bf16x16 BniT = load_bT(bi_, WD, ks, 0, KBIG,  scl);
    R = wmma_bf16(Anr,  BnrT, R);  R = wmma_bf16(Ani,  BniT, R);
    I = wmma_bf16(Anr,  BniT, I);  I = wmma_bf16(Anin, BnrT, I);
  }
  f32x8 f;
#pragma unroll
  for (int r = 0; r < 8; ++r) f[r] = fminf(R[r] * R[r] + I[r] * I[r], 1.f);
  store_c(fid + (size_t)b * WN * WN, WN, 0, 0, f);
}

// qkv GEMMs: (B*N,128)@(128,384), outputs stored bf16.
__global__ void k_qkv(const float* __restrict__ sr, const float* __restrict__ si,
                      const float* __restrict__ Wr, const float* __restrict__ br,
                      const float* __restrict__ Wi, const float* __restrict__ bi,
                      __bf16* __restrict__ qr, __bf16* __restrict__ qi) {
  const int gw = gwave();
  const int r0 = (gw / 24) * 16, n0 = (gw % 24) * 16;
  const int col = threadIdx.x & 15;
  f32x8 cr = {}, ci = {};
  for (int ks = 0; ks < WD; ks += 32) {
    cr = wmma_bf16(load_a(sr, WD, r0, ks, KBIG, 1.f), load_b(Wr, 384, ks, n0, KBIG, 1.f), cr);
    ci = wmma_bf16(load_a(si, WD, r0, ks, KBIG, 1.f), load_b(Wi, 384, ks, n0, KBIG, 1.f), ci);
  }
  const float vbr = br[n0 + col], vbi = bi[n0 + col];
#pragma unroll
  for (int r = 0; r < 8; ++r) { cr[r] += vbr; ci[r] += vbi; }
  store_c_bf(qr, 384, r0, n0, cr);
  store_c_bf(qi, 384, r0, n0, ci);
}

// per (b,h): complex scores -> magnitude softmax -> o = w @ v. All 16x16 WMMA.
__global__ void k_attn(const __bf16* __restrict__ qr, const __bf16* __restrict__ qi,
                       float* __restrict__ aor, float* __restrict__ aoi) {
  const int gw = gwave();
  const int b = gw >> 2, h = gw & 3;
  const int l = threadIdx.x & 31, col = l & 15, hi = l >> 4, w = threadIdx.x >> 5;
  const size_t rowbase = (size_t)b * WN;
  const int qoff = h * 32, koff = 128 + h * 32, voff = 256 + h * 32;
  bf16x16 Aqr  = load_a_bf(qr, 384, b * WN, qoff,  1.f);
  bf16x16 Aqi  = load_a_bf(qi, 384, b * WN, qoff,  1.f);
  bf16x16 Aqrn = load_a_bf(qr, 384, b * WN, qoff, -1.f);
  bf16x16 Bkr  = load_bT_bf(qr, 384, koff, b * WN);
  bf16x16 Bki  = load_bT_bf(qi, 384, koff, b * WN);
  f32x8 ar = {}, ai = {};
  ar = wmma_bf16(Aqr,  Bkr, ar);  ar = wmma_bf16(Aqi, Bki, ar);
  ai = wmma_bf16(Aqi,  Bkr, ai);  ai = wmma_bf16(Aqrn, Bki, ai);
  const float sc = 0.17677669529663687f;  // HD^-0.5
  __shared__ float wt[8][16][16];
#pragma unroll
  for (int r = 0; r < 8; ++r) {
    float xv = ar[r] * sc, yv = ai[r] * sc;
    wt[w][8 * hi + r][col] = sqrtf(xv * xv + yv * yv);
  }
  __syncthreads();
  if (l < 16) {  // row softmax over 16 keys
    float mx = -1e30f;
#pragma unroll
    for (int j = 0; j < 16; ++j) mx = fmaxf(mx, wt[w][l][j]);
    float e[16], s = 0.f;
#pragma unroll
    for (int j = 0; j < 16; ++j) { e[j] = __expf(wt[w][l][j] - mx); s += e[j]; }
    float inv = 1.f / s;
#pragma unroll
    for (int j = 0; j < 16; ++j) wt[w][l][j] = e[j] * inv;
  }
  __syncthreads();
  bf16x16 Aw;  // A = w, K=16 padded to 32
  {
    const int kof = hi << 3;
#pragma unroll
    for (int j = 0; j < 8; ++j) {
      Aw[j]     = (__bf16)wt[w][col][kof + j];
      Aw[8 + j] = (__bf16)0.f;
    }
  }
  const __bf16* qrb = qr + rowbase * 384;
  const __bf16* qib = qi + rowbase * 384;
  f32x8 z = {};
  f32x8 or0 = wmma_bf16(Aw, load_b_bf_k16(qrb, 384, voff),      z);
  f32x8 or1 = wmma_bf16(Aw, load_b_bf_k16(qrb, 384, voff + 16), z);
  f32x8 oi0 = wmma_bf16(Aw, load_b_bf_k16(qib, 384, voff),      z);
  f32x8 oi1 = wmma_bf16(Aw, load_b_bf_k16(qib, 384, voff + 16), z);
  store_c(aor + rowbase * WD, WD, 0, h * 32,      or0);
  store_c(aor + rowbase * WD, WD, 0, h * 32 + 16, or1);
  store_c(aoi + rowbase * WD, WD, 0, h * 32,      oi0);
  store_c(aoi + rowbase * WD, WD, 0, h * 32 + 16, oi1);
}

// att = o @ Wout + bout, plus column means over N (outputs 3 & 4).
__global__ void k_attout(const float* __restrict__ aor, const float* __restrict__ aoi,
                         const float* __restrict__ Wor, const float* __restrict__ bor,
                         const float* __restrict__ Woi, const float* __restrict__ boi,
                         float* __restrict__ attr, float* __restrict__ atti,
                         float* __restrict__ meanr, float* __restrict__ meani) {
  const int gw = gwave();
  const int b = gw >> 3, d0 = (gw & 7) * 16;
  const int l = threadIdx.x & 31, col = l & 15;
  f32x8 cr = {}, ci = {};
  for (int ks = 0; ks < WD; ks += 32) {
    bf16x16 arr = load_a(aor, WD, b * WN, ks, KBIG, 1.f);
    bf16x16 aii = load_a(aoi, WD, b * WN, ks, KBIG, 1.f);
    cr = wmma_bf16(arr, load_b(Wor, WD, ks, d0, KBIG, 1.f), cr);
    ci = wmma_bf16(aii, load_b(Woi, WD, ks, d0, KBIG, 1.f), ci);
  }
  const float vbr = bor[d0 + col], vbi = boi[d0 + col];
  float sumr = 0.f, sumi = 0.f;
#pragma unroll
  for (int r = 0; r < 8; ++r) { cr[r] += vbr; ci[r] += vbi; sumr += cr[r]; sumi += ci[r]; }
  store_c(attr + (size_t)b * WN * WD, WD, 0, d0, cr);
  store_c(atti + (size_t)b * WN * WD, WD, 0, d0, ci);
  sumr += __shfl_xor(sumr, 16);
  sumi += __shfl_xor(sumi, 16);
  if (l < 16) {
    meanr[(size_t)b * WD + d0 + l] = sumr * (1.f / 16.f);
    meani[(size_t)b * WD + d0 + l] = sumi * (1.f / 16.f);
  }
}

// measurement GEMMs: m(b,n,o) complex; store bf16; accumulate |m| sums per (b,n).
__global__ void k_measure1(const float* __restrict__ attr, const float* __restrict__ atti,
                           const float* __restrict__ Mr, const float* __restrict__ Mi,
                           __bf16* __restrict__ mrb, __bf16* __restrict__ mib,
                           float* __restrict__ magsum) {
  const int gw = gwave();
  const int ot = gw & 15, n = (gw >> 4) & 15, bt = gw >> 8;
  const int b0 = bt * 16, o0 = ot * 16;
  const int l = threadIdx.x & 31, col = l & 15, hi = l >> 4;
  f32x8 mr = {}, mi = {};
  const float* Ar_ = attr + n * WD;               // row stride 2048 selects fixed n
  const float* Ai_ = atti + n * WD;
  const float* Br_ = Mr + (size_t)n * WD * WO;
  const float* Bi_ = Mi + (size_t)n * WD * WO;
  for (int ks = 0; ks < WD; ks += 32) {
    bf16x16 Ar  = load_a(Ar_, WN * WD, b0, ks, KBIG,  1.f);
    bf16x16 Ai  = load_a(Ai_, WN * WD, b0, ks, KBIG,  1.f);
    bf16x16 Ain = load_a(Ai_, WN * WD, b0, ks, KBIG, -1.f);
    bf16x16 BMr = load_b(Br_, WO, ks, o0, KBIG, 1.f);
    bf16x16 BMi = load_b(Bi_, WO, ks, o0, KBIG, 1.f);
    mr = wmma_bf16(Ar,  BMr, mr);  mr = wmma_bf16(Ain, BMi, mr);
    mi = wmma_bf16(Ar,  BMi, mi);  mi = wmma_bf16(Ai,  BMr, mi);
  }
  float mag[8];
#pragma unroll
  for (int r = 0; r < 8; ++r) {
    int row = b0 + 8 * hi + r;
    size_t idx = ((size_t)(row * WN + n)) * WO + o0 + col;
    mrb[idx] = (__bf16)mr[r];
    mib[idx] = (__bf16)mi[r];
    mag[r] = sqrtf(mr[r] * mr[r] + mi[r] * mi[r]);
  }
#pragma unroll
  for (int r = 0; r < 8; ++r) {
    mag[r] += __shfl_xor(mag[r], 1);
    mag[r] += __shfl_xor(mag[r], 2);
    mag[r] += __shfl_xor(mag[r], 4);
    mag[r] += __shfl_xor(mag[r], 8);
  }
  if (col == 0) {
#pragma unroll
    for (int r = 0; r < 8; ++r)
      atomicAdd(&magsum[(b0 + 8 * hi + r) * WN + n], mag[r]);
  }
}

__global__ void k_probs(const float* __restrict__ magsum, float* __restrict__ probs) {
  int b = blockIdx.x * blockDim.x + threadIdx.x;
  float v[16], mx = -1e30f;
#pragma unroll
  for (int n = 0; n < 16; ++n) { v[n] = magsum[b * 16 + n] * (1.f / (float)WO); mx = fmaxf(mx, v[n]); }
  float s = 0.f;
#pragma unroll
  for (int n = 0; n < 16; ++n) { v[n] = __expf(v[n] - mx); s += v[n]; }
  float inv = 1.f / s;
#pragma unroll
  for (int n = 0; n < 16; ++n) probs[b * 16 + n] = v[n] * inv;
}

__global__ void k_measure2(const __bf16* __restrict__ mrb, const __bf16* __restrict__ mib,
                           const float* __restrict__ probs, float* __restrict__ classical) {
  int t = blockIdx.x * blockDim.x + threadIdx.x;
  int b = t >> 8, o = t & 255;
  float ar = 0.f, ai = 0.f;
  const float* pb = probs + b * 16;
#pragma unroll
  for (int n = 0; n < 16; ++n) {
    size_t idx = ((size_t)(b * 16 + n)) * WO + o;
    float p = pb[n];
    ar += p * (float)mrb[idx];
    ai += p * (float)mib[idx];
  }
  classical[t] = sqrtf(ar * ar + ai * ai);
}

// ---------------------------------------------------------------- launch
extern "C" void kernel_launch(void* const* d_in, const int* in_sizes, int n_in,
                              void* d_out, int out_size, void* d_ws, size_t ws_size,
                              hipStream_t stream) {
  (void)in_sizes; (void)n_in; (void)out_size; (void)ws_size;
  const float* x      = (const float*)d_in[0];
  const float* Wampr  = (const float*)d_in[1];
  const float* bampr  = (const float*)d_in[2];
  const float* Wampi  = (const float*)d_in[3];
  const float* bampi  = (const float*)d_in[4];
  const float* Wph    = (const float*)d_in[5];
  const float* bph    = (const float*)d_in[6];
  const float* Aparam = (const float*)d_in[7];
  const float* Wqkvr  = (const float*)d_in[8];
  const float* bqkvr  = (const float*)d_in[9];
  const float* Wqkvi  = (const float*)d_in[10];
  const float* bqkvi  = (const float*)d_in[11];
  const float* Woutr  = (const float*)d_in[12];
  const float* boutr  = (const float*)d_in[13];
  const float* Wouti  = (const float*)d_in[14];
  const float* bouti  = (const float*)d_in[15];
  const float* Mr     = (const float*)d_in[16];
  const float* Mi     = (const float*)d_in[17];

  float* out = (float*)d_out;
  float* out_classical = out;                // (B,O)   2,097,152
  float* out_probs     = out + 2097152;      // (B,N)     131,072
  float* out_meanr     = out + 2228224;      // (B,D)   1,048,576
  float* out_meani     = out + 3276800;      // (B,D)   1,048,576
  float* out_purity    = out + 4325376;      // scalar
  float* out_ent       = out + 4325377;      // scalar
  float* out_fid       = out + 4325378;      // (B,N,N) 2,097,152
  float* out_phvar     = out + 6422530;      // scalar

  char* wp = (char*)d_ws;
  auto alloc = [&](size_t bytes) -> char* {
    char* p = wp; wp += (bytes + 255) & ~(size_t)255; return p;
  };
  float*  Ur     = (float*)alloc(256 * 4);
  float*  Ui     = (float*)alloc(256 * 4);
  float*  phases = (float*)alloc((size_t)WB * WN * 4);
  float*  s_r    = (float*)alloc((size_t)WB * WN * WD * 4);   // reused as aor
  float*  s_i    = (float*)alloc((size_t)WB * WN * WD * 4);   // reused as aoi
  float*  invn   = (float*)alloc((size_t)WB * WN * 4);
  float*  isr    = (float*)alloc((size_t)WB * WN * WD * 4);
  float*  isi    = (float*)alloc((size_t)WB * WN * WD * 4);
  float*  fninv  = (float*)alloc((size_t)WB * WN * 4);
  float*  pbn    = (float*)alloc((size_t)WB * WN * 4);
  float*  attr   = (float*)alloc((size_t)WB * WN * WD * 4);
  float*  atti   = (float*)alloc((size_t)WB * WN * WD * 4);
  float*  magsum = (float*)alloc((size_t)WB * WN * 4);
  __bf16* qkvr   = (__bf16*)alloc((size_t)WB * WN * 384 * 2); // reused for m_r
  __bf16* qkvi   = (__bf16*)alloc((size_t)WB * WN * 384 * 2); // reused for m_i
  float*  aor    = s_r;     // s_r/s_i dead after k_interf
  float*  aoi    = s_i;
  __bf16* mrb    = qkvr;    // qkv dead after k_attn
  __bf16* mib    = qkvi;

  k_init        <<<  512, 256, 0, stream>>>(magsum, out_purity, out_ent, out_phvar);
  k_cayley      <<<    1,  32, 0, stream>>>(Aparam, Ur, Ui);
  k_phases      <<<   64, 256, 0, stream>>>(x, Wph, bph, phases, out_phvar);
  k_encode      <<< 8192, 256, 0, stream>>>(x, Wampr, bampr, Wampi, bampi, phases, s_r, s_i);
  k_norms       <<<16384, 256, 0, stream>>>(s_r, s_i, invn);
  k_interf      <<< 8192, 256, 0, stream>>>(Ur, Ui, s_r, s_i, invn, isr, isi);
  k_state_stats <<<16384, 256, 0, stream>>>(isr, isi, fninv, pbn);
  k_purity_final<<<   32, 256, 0, stream>>>(pbn, out_purity, out_ent);
  k_fidelity    <<< 1024, 256, 0, stream>>>(isr, isi, fninv, out_fid);
  k_qkv         <<<24576, 256, 0, stream>>>(isr, isi, Wqkvr, bqkvr, Wqkvi, bqkvi, qkvr, qkvi);
  k_attn        <<< 4096, 256, 0, stream>>>(qkvr, qkvi, aor, aoi);
  k_attout      <<< 8192, 256, 0, stream>>>(aor, aoi, Woutr, boutr, Wouti, bouti,
                                            attr, atti, out_meanr, out_meani);
  k_measure1    <<<16384, 256, 0, stream>>>(attr, atti, Mr, Mi, mrb, mib, magsum);
  k_probs       <<<   32, 256, 0, stream>>>(magsum, out_probs);
  k_measure2    <<< 8192, 256, 0, stream>>>(mrb, mib, out_probs, out_classical);
}
